// MLA_58832462021309
// MI455X (gfx1250) — compile-verified, matrix operand order checked by reading, not run on hardware
//
#include <hip/hip_runtime.h>
#include <hip/hip_bf16.h>

// ---------------------------------------------------------------------------
// MLA forward, CDNA5 (gfx1250) — all GEMMs via v_wmma_f32_16x16x32_bf16,
// GEMM tile staging via GLOBAL_LOAD_ASYNC_TO_LDS_B128 (ASYNCcnt path).
// B=2 S=2048 D=2048 H=16 C=512 DN=128 DR=64 DV=128
// ---------------------------------------------------------------------------

#define BB   2
#define SS   2048
#define DD   2048
#define HH   16
#define CC   512
#define DNN  128
#define DRR  64
#define DVV  128
#define BS   (BB * SS)            // 4096 tokens
#define QDIM (DNN + DRR)          // 192
#define CATD (CC + DRR)           // 576
#define KLD  584                  // padded LDS row pitch for 576-elem rows
#define SCALE_F 0.07216878364870323f   // (DN+DR)^-0.5 = 1/sqrt(192)
#define NEG_F  (-1.0e9f)

#define GLOBAL_AS __attribute__((address_space(1)))
#define LDS_AS    __attribute__((address_space(3)))

typedef __attribute__((ext_vector_type(16))) __bf16 v16bf;
typedef __attribute__((ext_vector_type(8)))  __bf16 v8bf;
typedef __attribute__((ext_vector_type(8)))  float  v8f;
typedef int v4i32 __attribute__((vector_size(16)));

__device__ inline __bf16 f2bf(float f) {
  union { float f; unsigned u; } x; x.f = f;
  unsigned r = x.u + 0x7FFFu + ((x.u >> 16) & 1u);   // round-to-nearest-even
  unsigned short h = (unsigned short)(r >> 16);
  return __builtin_bit_cast(__bf16, h);
}

__device__ inline v8f v8f_zero() {
  v8f z;
  #pragma unroll
  for (int i = 0; i < 8; ++i) z[i] = 0.f;
  return z;
}

// 16x32 bf16 fragment (A-layout; B(N,K) row-major uses the same pattern).
// lane 0-15: row=lane,    K = 0..7  and 16..23
// lane16-31: row=lane-16, K = 8..15 and 24..31
__device__ inline v16bf load_frag(const __bf16* p, int ld) {
  int lane = threadIdx.x & 31;
  int r    = lane & 15;
  int kh   = (lane & 16) ? 8 : 0;
  const __bf16* q = p + (size_t)r * ld + kh;
  v8bf lo = *(const v8bf*)q;
  v8bf hi = *(const v8bf*)(q + 16);
  v16bf f;
  #pragma unroll
  for (int i = 0; i < 8; ++i) { f[i] = lo[i]; f[8 + i] = hi[i]; }
  return f;
}

__device__ inline v8f wmma_bf16(v16bf a, v16bf b, v8f c) {
  return __builtin_amdgcn_wmma_f32_16x16x32_bf16(false, a, false, b,
                                                 (short)0, c, false, false);
}

// async global -> LDS, 16B per lane, tracked by ASYNCcnt
__device__ inline void async_b128(const __bf16* g, __bf16* l) {
  __builtin_amdgcn_global_load_async_to_lds_b128(
      (GLOBAL_AS v4i32*)g, (LDS_AS v4i32*)l, 0, 0);
}
__device__ inline void wait_async0() {
  asm volatile("s_wait_asynccnt 0x0" ::: "memory");
}

template <typename T> __device__ inline T cvt_out(float v);
template <> __device__ inline float  cvt_out<float >(float v) { return v; }
template <> __device__ inline __bf16 cvt_out<__bf16>(float v) { return f2bf(v); }

// ---------------------------------------------------------------------------
// fp32 -> bf16 cast
// ---------------------------------------------------------------------------
__global__ void cast_f32_bf16(const float* __restrict__ in,
                              __bf16* __restrict__ out, long n) {
  long i = (long)blockIdx.x * 256 + threadIdx.x;
  if (i < n) out[i] = f2bf(in[i]);
}

// wkcT[h][c][d] = wkv_b[h*256 + d][c]   (transpose w_kc into (N=C, K=DN))
__global__ void prep_wkcT(const float* __restrict__ wkv_b,
                          __bf16* __restrict__ wkcT) {
  long i = (long)blockIdx.x * 256 + threadIdx.x;
  if (i >= (long)HH * CC * DNN) return;
  int d = (int)(i & 127);
  long r = i >> 7;
  int c = (int)(r & 511);
  int h = (int)(r >> 9);
  wkcT[i] = f2bf(wkv_b[((size_t)(h * 256 + d)) * CC + c]);
}

// ---------------------------------------------------------------------------
// WMMA GEMM:  D[M,N] = A[M,K] @ B[N,K]^T  (bf16 in, fp32 acc, OutT out)
// block = 256 thr (8 waves), tile 128(M) x 64(N), K slab = 32 via async LDS.
// grid: (N/64, M/128, batch)
// ---------------------------------------------------------------------------
template <typename OutT>
__global__ __launch_bounds__(256)
void gemm_bf16(const __bf16* __restrict__ A, long aRow, long aBatch,
               const __bf16* __restrict__ B, long bRow, long bBatch,
               OutT* __restrict__ C, long cRow, long cBatch, int K) {
  __shared__ __align__(16) __bf16 As[128 * 40];
  __shared__ __align__(16) __bf16 Bs[64 * 40];

  int tid = threadIdx.x, lane = tid & 31, w = tid >> 5;
  int wm = w >> 1, wn = w & 1;
  int z = blockIdx.z;

  const __bf16* Ab = A + (size_t)z * aBatch + (size_t)blockIdx.y * 128 * aRow;
  const __bf16* Bb = B + (size_t)z * bBatch + (size_t)blockIdx.x * 64 * bRow;

  v8f acc[2][2];
  #pragma unroll
  for (int i = 0; i < 2; ++i)
    #pragma unroll
    for (int j = 0; j < 2; ++j) acc[i][j] = v8f_zero();

  int arow = tid >> 1, acol = (tid & 1) * 16;
  int brow = tid >> 2, bcol = (tid & 3) * 8;

  for (int k0 = 0; k0 < K; k0 += 32) {
    __syncthreads();
    // async memory -> LDS staging (no VGPR round trip), ASYNCcnt tracked
    async_b128(Ab + (size_t)arow * aRow + k0 + acol,     As + arow * 40 + acol);
    async_b128(Ab + (size_t)arow * aRow + k0 + acol + 8, As + arow * 40 + acol + 8);
    async_b128(Bb + (size_t)brow * bRow + k0 + bcol,     Bs + brow * 40 + bcol);
    if (k0 + 32 < K) {  // lowers to global_prefetch_b8 on gfx1250
      __builtin_prefetch(Ab + (size_t)arow * aRow + k0 + 32 + acol, 0, 1);
      __builtin_prefetch(Bb + (size_t)brow * bRow + k0 + 32 + bcol, 0, 1);
    }
    wait_async0();
    __syncthreads();
    v16bf a0 = load_frag(As + (wm * 32) * 40, 40);
    v16bf a1 = load_frag(As + (wm * 32 + 16) * 40, 40);
    v16bf b0 = load_frag(Bs + (wn * 32) * 40, 40);
    v16bf b1 = load_frag(Bs + (wn * 32 + 16) * 40, 40);
    acc[0][0] = wmma_bf16(a0, b0, acc[0][0]);
    acc[0][1] = wmma_bf16(a0, b1, acc[0][1]);
    acc[1][0] = wmma_bf16(a1, b0, acc[1][0]);
    acc[1][1] = wmma_bf16(a1, b1, acc[1][1]);
  }

  OutT* Cb = C + (size_t)z * cBatch + (size_t)blockIdx.y * 128 * cRow +
             (size_t)blockIdx.x * 64;
  int n = lane & 15, mh = (lane & 16) ? 8 : 0;
  #pragma unroll
  for (int mt = 0; mt < 2; ++mt)
    #pragma unroll
    for (int nt = 0; nt < 2; ++nt) {
      OutT* p = Cb + (size_t)(wm * 32 + mt * 16 + mh) * cRow +
                wn * 32 + nt * 16 + n;
      #pragma unroll
      for (int i = 0; i < 8; ++i)
        p[(size_t)i * cRow] = cvt_out<OutT>(acc[mt][nt][i]);
    }
}

// ---------------------------------------------------------------------------
// q epilogue: split q into q_nope (bf16) and RoPE'd q_pe -> qcat[...,512:576]
// one thread per (token, head)
// ---------------------------------------------------------------------------
__global__ void rope_q(const float* __restrict__ qf,
                       const int* __restrict__ start_pos,
                       __bf16* __restrict__ qn, __bf16* __restrict__ qcat) {
  int idx = blockIdx.x * blockDim.x + threadIdx.x;
  int t = idx >> 4, h = idx & 15;
  const float* src = qf + (size_t)t * (HH * QDIM) + h * QDIM;
  __bf16* np = qn + (size_t)t * (HH * DNN) + h * DNN;
  #pragma unroll 4
  for (int d = 0; d < DNN; ++d) np[d] = f2bf(src[d]);
  __bf16* pe = qcat + (size_t)t * (HH * CATD) + h * CATD + CC;
  int s = t & (SS - 1);
  float pos = (float)(*start_pos + s);
  for (int j = 0; j < DRR / 2; ++j) {
    float inv = __powf(10000.f, -((float)(2 * j)) / (float)DRR);
    float ang = pos * inv, c = __cosf(ang), sn = __sinf(ang);
    float x1 = src[DNN + 2 * j], x2 = src[DNN + 2 * j + 1];
    pe[2 * j]     = f2bf(x1 * c - x2 * sn);
    pe[2 * j + 1] = f2bf(x1 * sn + x2 * c);
  }
}

// ---------------------------------------------------------------------------
// kv epilogue: layernorm(kv_c) -> kcat[...,0:512]; RoPE(k_pe) -> kcat[...,512:]
// one block (128 thr) per token
// ---------------------------------------------------------------------------
__global__ void ln_rope_k(const float* __restrict__ kvf,
                          const float* __restrict__ g,
                          const float* __restrict__ bvec,
                          const int* __restrict__ start_pos,
                          __bf16* __restrict__ kcat) {
  int t = blockIdx.x, tid = threadIdx.x;
  const float* row = kvf + (size_t)t * CATD;
  __shared__ float red[128];

  float s = 0.f;
  for (int i = tid; i < CC; i += 128) s += row[i];
  red[tid] = s; __syncthreads();
  for (int off = 64; off > 0; off >>= 1) {
    if (tid < off) red[tid] += red[tid + off];
    __syncthreads();
  }
  float mean = red[0] / (float)CC;
  __syncthreads();

  float v = 0.f;
  for (int i = tid; i < CC; i += 128) { float d = row[i] - mean; v += d * d; }
  red[tid] = v; __syncthreads();
  for (int off = 64; off > 0; off >>= 1) {
    if (tid < off) red[tid] += red[tid + off];
    __syncthreads();
  }
  float rstd = rsqrtf(red[0] / (float)CC + 1e-5f);

  __bf16* out = kcat + (size_t)t * CATD;
  for (int i = tid; i < CC; i += 128)
    out[i] = f2bf((row[i] - mean) * rstd * g[i] + bvec[i]);

  if (tid < DRR / 2) {
    int sidx = t & (SS - 1);
    float pos = (float)(*start_pos + sidx);
    float inv = __powf(10000.f, -((float)(2 * tid)) / (float)DRR);
    float ang = pos * inv, c = __cosf(ang), sn = __sinf(ang);
    float x1 = row[CC + 2 * tid], x2 = row[CC + 2 * tid + 1];
    out[CC + 2 * tid]     = f2bf(x1 * c - x2 * sn);
    out[CC + 2 * tid + 1] = f2bf(x1 * sn + x2 * c);
  }
}

// ---------------------------------------------------------------------------
// Flash attention (absorbed form). Scores over 576-dim qcat/kcat, PV over
// 512-dim kv_c. 128 thr = 4 waves; wave owns 16 q rows; wave pairs split the
// 512-dim ctx accumulator (256 cols each, 128 acc VGPRs/lane).
//
// LDS-resident operands (fits 2 workgroups per 320KB WGP):
//   qtile : 32 q-rows x 576, loaded ONCE, reused for every key block
//   ktile : 32 keys  x 576, loaded once per key block (single global read)
//   kvcT  : kv_c^T 512 x 32, scattered during the same staging pass
//   Pls   : per-wave softmax'd P tiles (D-layout -> A-fragment round trip)
// grid: (S/32, H, B)
// ---------------------------------------------------------------------------
__global__ __launch_bounds__(128)
void mla_attn(const __bf16* __restrict__ qcat, const __bf16* __restrict__ kcat,
              __bf16* __restrict__ ctxb) {
  __shared__ __align__(16) __bf16 qtile[32 * KLD];     // 36.5 KB
  __shared__ __align__(16) __bf16 ktile[32 * KLD];     // 36.5 KB
  __shared__ __align__(16) __bf16 kvcT[CC * 40];       // 40 KB
  __shared__ __align__(16) __bf16 Pls[4 * 16 * 40];    // 5 KB

  int tid = threadIdx.x, lane = tid & 31, w = tid >> 5;
  int bz = blockIdx.z, h = blockIdx.y;
  int s0 = blockIdx.x * 32;
  int qm = (w >> 1) * 16;             // this wave's q-row offset in the tile
  int cHalf = (w & 1) * 256;          // this wave's ctx column half

  const __bf16* kbase = kcat + (size_t)(bz * SS) * CATD;

  // ---- stage Q tile once: 32 rows x 576 = 2304 v8 loads -------------------
  {
    const __bf16* qsrc =
        qcat + ((size_t)(bz * SS + s0)) * (HH * CATD) + (size_t)h * CATD;
    #pragma unroll
    for (int i = 0; i < 18; ++i) {
      int L = tid + 128 * i;          // 0..2303
      int row = L / 72;               // 72 v8-loads per 576-elem row
      int c8 = (L % 72) * 8;
      *(v8bf*)(qtile + row * KLD + c8) =
          *(const v8bf*)(qsrc + (size_t)row * (HH * CATD) + c8);
    }
  }

  float mi[8], li[8];
  v8f cacc[16];
  #pragma unroll
  for (int i = 0; i < 8; ++i) { mi[i] = -1e30f; li[i] = 0.f; }
  #pragma unroll
  for (int j = 0; j < 16; ++j) cacc[j] = v8f_zero();

  int n = lane & 15, mh = (lane & 16) ? 8 : 0;

  for (int t0 = 0; t0 < s0 + 32; t0 += 32) {
    __syncthreads();
    // ---- stage K block: ktile (row-major) + kvcT (transposed kv_c part) --
    const __bf16* ksrc = kbase + (size_t)t0 * CATD;
    #pragma unroll
    for (int i = 0; i < 18; ++i) {
      int L = tid + 128 * i;          // 0..2303
      int key = L / 72;
      int c8 = (L % 72) * 8;
      v8bf v = *(const v8bf*)(ksrc + (size_t)key * CATD + c8);
      *(v8bf*)(ktile + key * KLD + c8) = v;
      if (c8 < CC) {
        #pragma unroll
        for (int j = 0; j < 8; ++j) kvcT[(c8 + j) * 40 + key] = v[j];
      }
    }
    __syncthreads();

    // ---- scores: 16 q-rows x 32 keys, K = 576 (all operands in LDS) -----
    v8f a0 = v8f_zero(), a1 = v8f_zero();
    #pragma unroll
    for (int kk = 0; kk < CATD / 32; ++kk) {
      v16bf qfr = load_frag(qtile + qm * KLD + kk * 32, KLD);
      v16bf k0f = load_frag(ktile + kk * 32, KLD);
      v16bf k1f = load_frag(ktile + 16 * KLD + kk * 32, KLD);
      a0 = wmma_bf16(qfr, k0f, a0);
      a1 = wmma_bf16(qfr, k1f, a1);
    }

    // ---- online softmax (per-wave: all 8 rows live in 16-lane groups) ---
    float pa[8], pb[8];
    #pragma unroll
    for (int i = 0; i < 8; ++i) {
      int sq = s0 + qm + mh + i;
      float sa = a0[i] * SCALE_F; if (t0 + n > sq)       sa = NEG_F;
      float sb = a1[i] * SCALE_F; if (t0 + 16 + n > sq)  sb = NEG_F;
      float rm = fmaxf(sa, sb);
      rm = fmaxf(rm, __shfl_xor(rm, 1, 32));
      rm = fmaxf(rm, __shfl_xor(rm, 2, 32));
      rm = fmaxf(rm, __shfl_xor(rm, 4, 32));
      rm = fmaxf(rm, __shfl_xor(rm, 8, 32));
      float nm = fmaxf(mi[i], rm);
      pa[i] = __expf(sa - nm);
      pb[i] = __expf(sb - nm);
      float rs = pa[i] + pb[i];
      rs += __shfl_xor(rs, 1, 32);
      rs += __shfl_xor(rs, 2, 32);
      rs += __shfl_xor(rs, 4, 32);
      rs += __shfl_xor(rs, 8, 32);
      float corr = __expf(mi[i] - nm);
      li[i] = li[i] * corr + rs;
      mi[i] = nm;
      #pragma unroll
      for (int j = 0; j < 16; ++j) cacc[j][i] *= corr;
    }

    // ---- P -> LDS (D-layout -> A-fragment layout) -----------------------
    __bf16* Pw = Pls + w * 16 * 40;
    #pragma unroll
    for (int i = 0; i < 8; ++i) {
      Pw[(mh + i) * 40 + n]      = f2bf(pa[i]);
      Pw[(mh + i) * 40 + 16 + n] = f2bf(pb[i]);
    }
    __syncthreads();

    // ---- PV: ctx += P(16x32) @ kv_c(32x512) over this wave's 256 cols ---
    v16bf pf = load_frag(Pw, 40);
    #pragma unroll
    for (int j = 0; j < 16; ++j) {
      v16bf bf_ = load_frag(kvcT + (size_t)(cHalf + j * 16) * 40, 40);
      cacc[j] = wmma_bf16(pf, bf_, cacc[j]);
    }
  }

  // ---- epilogue: ctx / l -> bf16 ----------------------------------------
  #pragma unroll
  for (int i = 0; i < 8; ++i) {
    float r = 1.f / li[i];
    size_t tok = (size_t)(bz * SS) + s0 + qm + mh + i;
    __bf16* op = ctxb + (tok * HH + h) * CC + cHalf;
    #pragma unroll
    for (int j = 0; j < 16; ++j) op[j * 16 + n] = f2bf(cacc[j][i] * r);
  }
}

// ---------------------------------------------------------------------------
// Host launcher
// ---------------------------------------------------------------------------
extern "C" void kernel_launch(void* const* d_in, const int* in_sizes, int n_in,
                              void* d_out, int out_size, void* d_ws,
                              size_t ws_size, hipStream_t stream) {
  const float* x      = (const float*)d_in[0];
  const float* wq     = (const float*)d_in[1];
  const float* wkv_a  = (const float*)d_in[2];
  const float* kv_g   = (const float*)d_in[3];
  const float* kv_b   = (const float*)d_in[4];
  const float* wkv_b  = (const float*)d_in[5];
  const float* wo     = (const float*)d_in[6];
  const int*   sp     = (const int*)d_in[7];
  float* out          = (float*)d_out;

  char* p = (char*)d_ws;
  auto alloc = [&](size_t bytes) -> void* {
    void* r = (void*)p;
    p += (bytes + 255) & ~(size_t)255;
    return r;
  };

  __bf16* x_bf    = (__bf16*)alloc((size_t)BS * DD * 2);
  __bf16* wq_bf   = (__bf16*)alloc((size_t)HH * QDIM * DD * 2);
  __bf16* wkva_bf = (__bf16*)alloc((size_t)CATD * DD * 2);
  __bf16* wkvb_bf = (__bf16*)alloc((size_t)HH * 256 * CC * 2);
  __bf16* wo_bf   = (__bf16*)alloc((size_t)DD * DD * 2);
  __bf16* wkcT    = (__bf16*)alloc((size_t)HH * CC * DNN * 2);
  float*  qf      = (float*) alloc((size_t)BS * HH * QDIM * 4);
  float*  kvf     = (float*) alloc((size_t)BS * CATD * 4);
  __bf16* qn      = (__bf16*)alloc((size_t)BS * HH * DNN * 2);
  __bf16* qcat    = (__bf16*)alloc((size_t)BS * HH * CATD * 2);
  __bf16* kcat    = (__bf16*)alloc((size_t)BS * CATD * 2);
  __bf16* ctxb    = (__bf16*)alloc((size_t)BS * HH * CC * 2);
  __bf16* hout    = (__bf16*)alloc((size_t)BS * HH * DVV * 2);

  auto cast = [&](const float* src, __bf16* dst, long cnt) {
    cast_f32_bf16<<<dim3((unsigned)((cnt + 255) / 256)), 256, 0, stream>>>(
        src, dst, cnt);
  };
  cast(x,     x_bf,    (long)BS * DD);
  cast(wq,    wq_bf,   (long)HH * QDIM * DD);
  cast(wkv_a, wkva_bf, (long)CATD * DD);
  cast(wkv_b, wkvb_bf, (long)HH * 256 * CC);
  cast(wo,    wo_bf,   (long)DD * DD);
  prep_wkcT<<<dim3((HH * CC * DNN + 255) / 256), 256, 0, stream>>>(wkv_b, wkcT);

  // 1) q = x @ wq^T   (4096 x 3072, K=2048) -> fp32
  gemm_bf16<float><<<dim3(HH * QDIM / 64, BS / 128, 1), 256, 0, stream>>>(
      x_bf, DD, 0, wq_bf, DD, 0, qf, (long)HH * QDIM, 0, DD);

  // 2) kvf = x @ wkv_a^T  (4096 x 576, K=2048) -> fp32
  gemm_bf16<float><<<dim3(CATD / 64, BS / 128, 1), 256, 0, stream>>>(
      x_bf, DD, 0, wkva_bf, DD, 0, kvf, CATD, 0, DD);

  // 3) epilogues: split/RoPE q, LN+RoPE kv
  rope_q<<<dim3(BS * HH / 128), 128, 0, stream>>>(qf, sp, qn, qcat);
  ln_rope_k<<<dim3(BS), 128, 0, stream>>>(kvf, kv_g, kv_b, sp, kcat);

  // 4) q_abs[h] = q_nope[h] @ w_kc[h]  (per head: 4096 x 512, K=128) -> qcat
  gemm_bf16<__bf16><<<dim3(CC / 64, BS / 128, HH), 256, 0, stream>>>(
      qn, (long)HH * DNN, DNN,
      wkcT, DNN, (long)CC * DNN,
      qcat, (long)HH * CATD, CATD, DNN);

  // 5) flash attention -> ctx (BS, H, 512) bf16
  mla_attn<<<dim3(SS / 32, HH, BB), 128, 0, stream>>>(qcat, kcat, ctxb);

  // 6) hout[h] = ctx[h] @ w_vc[h]^T  (per head: 4096 x 128, K=512)
  gemm_bf16<__bf16><<<dim3(DVV / 64, BS / 128, HH), 256, 0, stream>>>(
      ctxb, (long)HH * CC, CC,
      wkvb_bf + (size_t)DNN * CC, CC, (long)256 * CC,
      hout, (long)HH * DVV, DVV, CC);

  // 7) out = hout @ wo^T  (4096 x 2048, K=2048) -> fp32
  gemm_bf16<float><<<dim3(DD / 64, BS / 128, 1), 256, 0, stream>>>(
      hout, (long)HH * DVV, 0, wo_bf, (long)HH * DVV, 0, out, DD, 0,
      (long)HH * DVV);
}
